// MultiHeadSelfAttention_24249385353799
// MI455X (gfx1250) — matrix operand order repaired and checked by reading, not compiled, not run on hardware
//
#include <hip/hip_runtime.h>
#include <hip/hip_bf16.h>

// ---------------------------------------------------------------------------
// MultiHeadSelfAttention forward for MI455X (gfx1250), bf16 WMMA + flash attn
// B=2, S=4096, D=512, H=8, DK=64
// ---------------------------------------------------------------------------

#define Bn 2
#define Sn 4096
#define Dn 512
#define Hn 8
#define DKn 64
#define Mrows (Bn * Sn)          // 8192 token rows

typedef __attribute__((ext_vector_type(16))) __bf16 v16bf;
typedef __attribute__((ext_vector_type(8)))  __bf16 v8bf;
typedef __attribute__((ext_vector_type(4)))  __bf16 v4bf;
typedef __attribute__((ext_vector_type(8)))  float  v8f;

__device__ __forceinline__ v16bf join16(v8bf lo, v8bf hi) {
    v16bf r;
#pragma unroll
    for (int j = 0; j < 8; ++j) { r[j] = lo[j]; r[8 + j] = hi[j]; }
    return r;
}

// A-fragment (16x32 bf16): lane row = lane&15, kbase = (lane>>4)*8.
// a[0..7]  <- rowptr[2*kbase + j]
// a[8..15] <- rowptr[2*kbase + 8 + j]
__device__ __forceinline__ v16bf load_afrag(const __bf16* rowptr, int kbase) {
    v8bf lo = *(const v8bf*)(rowptr + 2 * kbase);
    v8bf hi = *(const v8bf*)(rowptr + 2 * kbase + 8);
    return join16(lo, hi);
}

// B-fragment (32x16 bf16): lane col = lane&15, K = (lane>>4)*16 + j,
// 16 contiguous elements at p.
__device__ __forceinline__ v16bf load_bfrag(const __bf16* p) {
    v8bf lo = *(const v8bf*)(p);
    v8bf hi = *(const v8bf*)(p + 8);
    return join16(lo, hi);
}

__device__ __forceinline__ v8f wmma_bf16(v16bf a, v16bf b, v8f c) {
    return __builtin_amdgcn_wmma_f32_16x16x32_bf16(
        /*neg_a=*/false, a, /*neg_b=*/false, b,
        /*c_mod=*/(short)0, c, /*reuse_a=*/false, /*reuse_b=*/false);
}

// ---------------------------------------------------------------------------
// Kernel 1: fp32 -> bf16 cast (vectorized)
// ---------------------------------------------------------------------------
__global__ __launch_bounds__(256) void k_cast_bf16(const float* __restrict__ src,
                                                   __bf16* __restrict__ dst, int n4) {
    int t = blockIdx.x * blockDim.x + threadIdx.x;
    if (t < n4) {
        float4 f = ((const float4*)src)[t];
        v4bf o;
        o[0] = (__bf16)f.x; o[1] = (__bf16)f.y;
        o[2] = (__bf16)f.z; o[3] = (__bf16)f.w;
        ((v4bf*)dst)[t] = o;
    }
}

// ---------------------------------------------------------------------------
// Kernel 2: pack all 4 weight matrices (512x512 fp32, (in,out)) into
// B-fragment-friendly bf16 layout:
//   out[ w*2^18 + ((kblk*512 + n)*2 + half)*16 + j ] = W[(kblk*32+half*16+j)*512 + n]
// ---------------------------------------------------------------------------
__global__ __launch_bounds__(256) void k_pack_w(const float* __restrict__ W0,
                                                const float* __restrict__ W1,
                                                const float* __restrict__ W2,
                                                const float* __restrict__ W3,
                                                __bf16* __restrict__ out) {
    int t = blockIdx.x * blockDim.x + threadIdx.x;       // 4 * 262144 threads
    int w = t >> 18;
    int r = t & 262143;
    int j    = r & 15;
    int half = (r >> 4) & 1;
    int n    = (r >> 5) & 511;
    int kblk = r >> 14;
    int K = kblk * 32 + half * 16 + j;
    const float* W = (w == 0) ? W0 : (w == 1) ? W1 : (w == 2) ? W2 : W3;
    out[t] = (__bf16)W[K * Dn + n];
}

// ---------------------------------------------------------------------------
// Kernel 3: fused Q/K/V projection. One wave = 16 rows x 64 cols (= one head's
// dk range) for all three outputs. Q scaled by 1/sqrt(DK). V stored transposed.
// ---------------------------------------------------------------------------
__global__ __launch_bounds__(128) void k_qkv(const __bf16* __restrict__ xb,
                                             const __bf16* __restrict__ wq,
                                             const __bf16* __restrict__ wk,
                                             const __bf16* __restrict__ wv,
                                             const float* __restrict__ bq,
                                             const float* __restrict__ bk,
                                             const float* __restrict__ bv,
                                             __bf16* __restrict__ qo,
                                             __bf16* __restrict__ ko,
                                             __bf16* __restrict__ vto) {
    const int lane  = threadIdx.x & 31;
    const int wave  = threadIdx.x >> 5;
    const int task  = blockIdx.x * 4 + wave;     // 4096 tasks
    const int mtile = task >> 3;                 // 0..511
    const int nch   = task & 7;                  // head index, 0..7
    const int l15   = lane & 15;
    const int hv    = lane >> 4;
    const int kbase = hv * 8;

    const __bf16* arow = xb + (size_t)(mtile * 16 + l15) * Dn;

    v8f aq[4], ak[4], av[4];
#pragma unroll
    for (int t = 0; t < 4; ++t) { aq[t] = {}; ak[t] = {}; av[t] = {}; }

    for (int k0 = 0; k0 < Dn; k0 += 32) {
        v16bf af = load_afrag(arow + k0, kbase);
        int kblk = k0 >> 5;
#pragma unroll
        for (int t = 0; t < 4; ++t) {
            int n = nch * 64 + t * 16 + l15;
            size_t off = ((size_t)(kblk * Dn + n) * 2 + hv) * 16;
            aq[t] = wmma_bf16(af, load_bfrag(wq + off), aq[t]);
            ak[t] = wmma_bf16(af, load_bfrag(wk + off), ak[t]);
            av[t] = wmma_bf16(af, load_bfrag(wv + off), av[t]);
        }
    }

    const int h = nch;
#pragma unroll
    for (int t = 0; t < 4; ++t) {
        int dk = t * 16 + l15;
        int c  = nch * 64 + dk;
        float bqv = bq[c], bkv = bk[c], bvv = bv[c];
#pragma unroll
        for (int i = 0; i < 8; ++i) {
            int r = mtile * 16 + hv * 8 + i;
            int b = r >> 12;            // /S
            int s = r & (Sn - 1);
            size_t hb = (size_t)(b * Hn + h);
            qo[(hb * Sn + s) * DKn + dk]  = (__bf16)((aq[t][i] + bqv) * 0.125f);
            ko[(hb * Sn + s) * DKn + dk]  = (__bf16)(ak[t][i] + bkv);
            vto[(hb * DKn + dk) * Sn + s] = (__bf16)(av[t][i] + bvv);
        }
    }
}

// ---------------------------------------------------------------------------
// Kernel 4: flash attention. One wave per (b, h, 16-row q-tile); streams K/V in
// 32-key chunks; online softmax; P transposed C-layout -> A-layout via LDS.
// ---------------------------------------------------------------------------
__global__ __launch_bounds__(256) void k_attn(const __bf16* __restrict__ qb,
                                              const __bf16* __restrict__ kb,
                                              const __bf16* __restrict__ vtb,
                                              const int* __restrict__ mask,
                                              __bf16* __restrict__ ctx) {
    __shared__ __bf16 sp[8][16 * 32];   // per-wave P staging (16 rows x 32 keys)

    const int lane = threadIdx.x & 31;
    const int wave = threadIdx.x >> 5;
    const int task = blockIdx.x * 8 + wave;          // 4096 tasks
    const int qt = task & 255;
    const int h  = (task >> 8) & 7;
    const int b  = task >> 11;
    const int l15 = lane & 15;
    const int hv  = lane >> 4;
    const int kbase = hv * 8;
    const int kb16  = hv * 16;

    const size_t headoff = (size_t)(b * Hn + h) * Sn * DKn;
    const __bf16* qrow   = qb + headoff + (size_t)(qt * 16 + l15) * DKn;
    const __bf16* khead  = kb + headoff;
    const __bf16* vthead = vtb + (size_t)(b * Hn + h) * DKn * Sn;
    const int* mrow = mask + b * Sn;

    v16bf qf0 = load_afrag(qrow, kbase);        // dk 0..31
    v16bf qf1 = load_afrag(qrow + 32, kbase);   // dk 32..63

    float mi[8], li[8];
    v8f acc[4];
#pragma unroll
    for (int i = 0; i < 8; ++i) { mi[i] = -1e30f; li[i] = 0.0f; }
#pragma unroll
    for (int t = 0; t < 4; ++t) acc[t] = {};

    __bf16* myp = &sp[wave][0];

    for (int k0 = 0; k0 < Sn; k0 += 32) {
        const __bf16* kr0 = khead + (size_t)(k0 + l15) * DKn;
        const __bf16* kr1 = khead + (size_t)(k0 + 16 + l15) * DKn;

        v8f s0 = {}, s1 = {};
        s0 = wmma_bf16(qf0, load_bfrag(kr0 + kb16), s0);
        s0 = wmma_bf16(qf1, load_bfrag(kr0 + 32 + kb16), s0);
        s1 = wmma_bf16(qf0, load_bfrag(kr1 + kb16), s1);
        s1 = wmma_bf16(qf1, load_bfrag(kr1 + 32 + kb16), s1);

        if (mrow[k0 + l15] == 0) {
#pragma unroll
            for (int i = 0; i < 8; ++i) s0[i] = -1e9f;
        }
        if (mrow[k0 + 16 + l15] == 0) {
#pragma unroll
            for (int i = 0; i < 8; ++i) s1[i] = -1e9f;
        }

        float p0[8], p1[8], alpha[8];
#pragma unroll
        for (int i = 0; i < 8; ++i) {
            float cm = fmaxf(s0[i], s1[i]);
            cm = fmaxf(cm, __shfl_xor(cm, 1, 32));
            cm = fmaxf(cm, __shfl_xor(cm, 2, 32));
            cm = fmaxf(cm, __shfl_xor(cm, 4, 32));
            cm = fmaxf(cm, __shfl_xor(cm, 8, 32));
            float mn = fmaxf(mi[i], cm);
            float al = __expf(mi[i] - mn);
            float e0 = __expf(s0[i] - mn);
            float e1 = __expf(s1[i] - mn);
            float rs = e0 + e1;
            rs += __shfl_xor(rs, 1, 32);
            rs += __shfl_xor(rs, 2, 32);
            rs += __shfl_xor(rs, 4, 32);
            rs += __shfl_xor(rs, 8, 32);
            li[i] = li[i] * al + rs;
            mi[i] = mn;
            alpha[i] = al; p0[i] = e0; p1[i] = e1;
        }
#pragma unroll
        for (int t = 0; t < 4; ++t)
#pragma unroll
            for (int i = 0; i < 8; ++i) acc[t][i] *= alpha[i];

        // C-layout -> LDS (row-major 16x32 bf16)
#pragma unroll
        for (int i = 0; i < 8; ++i) {
            int m = hv * 8 + i;
            myp[m * 32 + l15]      = (__bf16)p0[i];
            myp[m * 32 + 16 + l15] = (__bf16)p1[i];
        }
        asm volatile("s_wait_dscnt 0" ::: "memory");

        // LDS -> A-fragment of P (16 q-rows x 32 keys)
        v8bf plo = *(const v8bf*)(myp + l15 * 32 + kb16);
        v8bf phi = *(const v8bf*)(myp + l15 * 32 + kb16 + 8);
        v16bf pf = join16(plo, phi);

#pragma unroll
        for (int t = 0; t < 4; ++t) {
            const __bf16* vr = vthead + (size_t)(t * 16 + l15) * Sn + k0 + kb16;
            acc[t] = wmma_bf16(pf, load_bfrag(vr), acc[t]);
        }
    }

#pragma unroll
    for (int t = 0; t < 4; ++t) {
        int dk = t * 16 + l15;
#pragma unroll
        for (int i = 0; i < 8; ++i) {
            float o = acc[t][i] / li[i];
            int r = b * Sn + qt * 16 + hv * 8 + i;
            ctx[(size_t)r * Dn + h * 64 + dk] = (__bf16)o;
        }
    }
}

// ---------------------------------------------------------------------------
// Kernel 5: output projection ctx @ Wo + bo -> fp32 out
// ---------------------------------------------------------------------------
__global__ __launch_bounds__(128) void k_oproj(const __bf16* __restrict__ ctxb,
                                               const __bf16* __restrict__ wo,
                                               const float* __restrict__ bo,
                                               float* __restrict__ out) {
    const int lane  = threadIdx.x & 31;
    const int wave  = threadIdx.x >> 5;
    const int task  = blockIdx.x * 4 + wave;
    const int mtile = task >> 3;
    const int nch   = task & 7;
    const int l15   = lane & 15;
    const int hv    = lane >> 4;
    const int kbase = hv * 8;

    const __bf16* arow = ctxb + (size_t)(mtile * 16 + l15) * Dn;

    v8f acc[4];
#pragma unroll
    for (int t = 0; t < 4; ++t) acc[t] = {};

    for (int k0 = 0; k0 < Dn; k0 += 32) {
        v16bf af = load_afrag(arow + k0, kbase);
        int kblk = k0 >> 5;
#pragma unroll
        for (int t = 0; t < 4; ++t) {
            int n = nch * 64 + t * 16 + l15;
            size_t off = ((size_t)(kblk * Dn + n) * 2 + hv) * 16;
            acc[t] = wmma_bf16(af, load_bfrag(wo + off), acc[t]);
        }
    }

#pragma unroll
    for (int t = 0; t < 4; ++t) {
        int c = nch * 64 + t * 16 + l15;
        float bv = bo[c];
#pragma unroll
        for (int i = 0; i < 8; ++i) {
            int r = mtile * 16 + hv * 8 + i;
            out[(size_t)r * Dn + c] = acc[t][i] + bv;
        }
    }
}

// ---------------------------------------------------------------------------
// Host-side launch
// ---------------------------------------------------------------------------
extern "C" void kernel_launch(void* const* d_in, const int* in_sizes, int n_in,
                              void* d_out, int out_size, void* d_ws, size_t ws_size,
                              hipStream_t stream) {
    (void)in_sizes; (void)n_in; (void)out_size; (void)ws_size;

    const float* x    = (const float*)d_in[0];
    const int*   mask = (const int*)d_in[1];
    const float* Wq   = (const float*)d_in[2];
    const float* bq   = (const float*)d_in[3];
    const float* Wk   = (const float*)d_in[4];
    const float* bk   = (const float*)d_in[5];
    const float* Wv   = (const float*)d_in[6];
    const float* bv   = (const float*)d_in[7];
    const float* Wo   = (const float*)d_in[8];
    const float* bo   = (const float*)d_in[9];

    // workspace layout (bf16 elements)
    const size_t XE = (size_t)Mrows * Dn;       // 4,194,304
    const size_t WE = (size_t)Dn * Dn;          //   262,144 per matrix
    __bf16* xb   = (__bf16*)d_ws;
    __bf16* wpk  = xb + XE;                     // 4 packed matrices, contiguous
    __bf16* qbf  = wpk + 4 * WE;
    __bf16* kbf  = qbf + XE;
    __bf16* vtbf = kbf + XE;
    __bf16* ctxb = vtbf + XE;

    // 1) cast x to bf16
    k_cast_bf16<<<4096, 256, 0, stream>>>(x, xb, (int)(XE / 4));

    // 2) pack Wq, Wk, Wv, Wo
    k_pack_w<<<4096, 256, 0, stream>>>(Wq, Wk, Wv, Wo, wpk);

    // 3) fused QKV projection (Q pre-scaled, V transposed)
    k_qkv<<<1024, 128, 0, stream>>>(xb, wpk, wpk + WE, wpk + 2 * WE,
                                    bq, bk, bv, qbf, kbf, vtbf);

    // 4) flash attention
    k_attn<<<512, 256, 0, stream>>>(qbf, kbf, vtbf, mask, ctxb);

    // 5) output projection
    k_oproj<<<1024, 128, 0, stream>>>(ctxb, wpk + 3 * WE, bo, (float*)d_out);
}
